// LSTM_12171937317002
// MI455X (gfx1250) — compile-verified, matrix operand order checked by reading, not run on hardware
//
#include <hip/hip_runtime.h>
#include <hip/hip_bf16.h>

// ---------------------------------------------------------------------------
// LayerNorm-LSTM forward on gfx1250 (MI455X), bf16x3-split WMMA.
// S=512, B=64, I=512, H=1024, 4H=4096.
// ---------------------------------------------------------------------------

#define S_LEN 512
#define BATCH 64
#define IDIM  512
#define HDIM  1024
#define G4H   4096   // 4*H

typedef __bf16 bf16_t;
typedef __attribute__((ext_vector_type(16))) __bf16 v16bf;
typedef __attribute__((ext_vector_type(8)))  float  v8f;

__device__ __forceinline__ v8f wmma_bf16(v16bf a, v16bf b, v8f c) {
    // D = A(16x32 bf16) * B(32x16 bf16) + C(16x16 f32)
    return __builtin_amdgcn_wmma_f32_16x16x32_bf16(
        false, a, false, b, (short)0, c, false, false);
}

// Load one 16x32 bf16 fragment (A-style; also used for B via W rows = B^T,
// since W is stored row-major with K contiguous).
// Lane L: row = row0 + (L&15); holds K chunks [g*8, g*8+8) and [16+g*8, ...),
// g = L>>4  (matches ISA 7.12.2 16-bit A layout).
__device__ __forceinline__ v16bf load_frag(const bf16_t* __restrict__ base,
                                           int row0, int ld, int k0) {
    const int lane = threadIdx.x & 31;
    const int r = lane & 15;
    const int g = lane >> 4;
    const bf16_t* p = base + (size_t)(row0 + r) * (size_t)ld + k0 + g * 8;
    union { uint4 u[2]; v16bf v; } f;
    f.u[0] = *(const uint4*)(p);
    f.u[1] = *(const uint4*)(p + 16);
    return f.v;
}

__device__ __forceinline__ void store_tile(float* __restrict__ out, int ldn,
                                           int m0, int n0, v8f acc) {
    const int lane = threadIdx.x & 31;
    const int n  = lane & 15;
    const int mg = lane >> 4;
#pragma unroll
    for (int v = 0; v < 8; ++v)
        out[(size_t)(m0 + mg * 8 + v) * (size_t)ldn + n0 + n] = acc[v];
}

__device__ __forceinline__ float sigm(float x) { return 1.0f / (1.0f + expf(-x)); }

// --- split fp32 -> bf16 hi + bf16 lo (a ~= hi + lo) -------------------------
__global__ __launch_bounds__(256) void split_kernel(const float* __restrict__ src,
                                                    bf16_t* __restrict__ hi,
                                                    bf16_t* __restrict__ lo, int n) {
    int i = blockIdx.x * 256 + threadIdx.x;
    if (i < n) {
        float a = src[i];
        bf16_t h = (bf16_t)a;
        hi[i] = h;
        lo[i] = (bf16_t)(a - (float)h);
    }
}

// --- zero h/c state ---------------------------------------------------------
__global__ __launch_bounds__(256) void init_state_kernel(float* __restrict__ h,
                                                         float* __restrict__ c,
                                                         bf16_t* __restrict__ hh,
                                                         bf16_t* __restrict__ hl) {
    int i = blockIdx.x * 256 + threadIdx.x;   // BATCH*HDIM threads
    h[i] = 0.0f; c[i] = 0.0f;
    hh[i] = (bf16_t)0.0f; hl[i] = (bf16_t)0.0f;
}

// --- big parallel GEMM: Xg[SB x 4H] = x[SB x I] * Wx^T  (bf16x3) ------------
// grid = (SB/16, 4H/256), block = 128 (4 waves). Wave w: 16x64 tile.
__global__ __launch_bounds__(128) void gemm_x_kernel(const bf16_t* __restrict__ xh,
                                                     const bf16_t* __restrict__ xl,
                                                     const bf16_t* __restrict__ wh,
                                                     const bf16_t* __restrict__ wl,
                                                     float* __restrict__ Xg) {
    const int w  = threadIdx.x >> 5;
    const int m0 = blockIdx.x * 16;
    const int n0 = (blockIdx.y * 4 + w) * 64;
    v8f acc[4] = {};
#pragma unroll 1
    for (int k0 = 0; k0 < IDIM; k0 += 32) {
        v16bf ah = load_frag(xh, m0, IDIM, k0);
        v16bf al = load_frag(xl, m0, IDIM, k0);
#pragma unroll
        for (int t = 0; t < 4; ++t) {
            v16bf bh = load_frag(wh, n0 + t * 16, IDIM, k0);
            v16bf bl = load_frag(wl, n0 + t * 16, IDIM, k0);
            acc[t] = wmma_bf16(ah, bh, acc[t]);
            acc[t] = wmma_bf16(ah, bl, acc[t]);
            acc[t] = wmma_bf16(al, bh, acc[t]);
        }
    }
#pragma unroll
    for (int t = 0; t < 4; ++t)
        store_tile(Xg, G4H, m0, n0 + t * 16, acc[t]);
}

// --- per-step recurrent GEMM: Hg[B x 4H] = h[B x H] * Wh^T  (bf16x3) --------
// 512 waves: grid = 128 blocks x 128 thr. Wave: 16x32 tile, K=1024.
__global__ __launch_bounds__(128) void gemm_h_kernel(const bf16_t* __restrict__ hh,
                                                     const bf16_t* __restrict__ hl,
                                                     const bf16_t* __restrict__ wh,
                                                     const bf16_t* __restrict__ wl,
                                                     float* __restrict__ Hg) {
    const int W  = blockIdx.x * 4 + (threadIdx.x >> 5);
    const int m0 = (W & 3) * 16;        // 4 M-tiles (B=64)
    const int n0 = (W >> 2) * 32;       // 128 N-chunks of 32
    v8f acc[2] = {};
#pragma unroll 1
    for (int k0 = 0; k0 < HDIM; k0 += 32) {
        v16bf ah = load_frag(hh, m0, HDIM, k0);
        v16bf al = load_frag(hl, m0, HDIM, k0);
#pragma unroll
        for (int t = 0; t < 2; ++t) {
            v16bf bh = load_frag(wh, n0 + t * 16, HDIM, k0);
            v16bf bl = load_frag(wl, n0 + t * 16, HDIM, k0);
            acc[t] = wmma_bf16(ah, bh, acc[t]);
            acc[t] = wmma_bf16(ah, bl, acc[t]);
            acc[t] = wmma_bf16(al, bh, acc[t]);
        }
    }
#pragma unroll
    for (int t = 0; t < 2; ++t)
        store_tile(Hg, G4H, m0, n0 + t * 16, acc[t]);
}

// --- fused LSTM cell: LN over 4096 gates, gate math, LN over c, emit h ------
// One block (256 thr) per batch row. Thread t owns h-indices t*4..t*4+3.
__global__ __launch_bounds__(256) void cell_kernel(const float* __restrict__ Xg_s,
                                                   const float* __restrict__ Hg,
                                                   const float* __restrict__ bias,
                                                   const float* __restrict__ gg,
                                                   const float* __restrict__ gb,
                                                   const float* __restrict__ cg,
                                                   const float* __restrict__ cb,
                                                   float* __restrict__ h_state,
                                                   float* __restrict__ c_state,
                                                   bf16_t* __restrict__ h_hi,
                                                   bf16_t* __restrict__ h_lo) {
    const int b = blockIdx.x;
    const int t = threadIdx.x;
    __shared__ float s1[256], s2[256];

    const float* xr = Xg_s + (size_t)b * G4H;
    const float* hr = Hg   + (size_t)b * G4H;

    float z[4][4];
    float lsum = 0.f, lsq = 0.f;
#pragma unroll
    for (int g = 0; g < 4; ++g)
#pragma unroll
        for (int q = 0; q < 4; ++q) {
            int n = g * HDIM + t * 4 + q;
            float v = xr[n] + hr[n] + bias[n];
            z[g][q] = v; lsum += v; lsq += v * v;
        }
    s1[t] = lsum; s2[t] = lsq; __syncthreads();
    for (int off = 128; off > 0; off >>= 1) {
        if (t < off) { s1[t] += s1[t + off]; s2[t] += s2[t + off]; }
        __syncthreads();
    }
    float mean = s1[0] * (1.0f / G4H);
    float var  = s2[0] * (1.0f / G4H) - mean * mean;
    float rs   = rsqrtf(var + 1e-5f);
    __syncthreads();

    float cn[4], ov[4];
    float csum = 0.f, csq = 0.f;
#pragma unroll
    for (int q = 0; q < 4; ++q) {
        int h = t * 4 + q;
        float iv = (z[0][q] - mean) * rs * gg[h]            + gb[h];
        float fv = (z[1][q] - mean) * rs * gg[HDIM + h]     + gb[HDIM + h];
        float jv = (z[2][q] - mean) * rs * gg[2 * HDIM + h] + gb[2 * HDIM + h];
        float oo = (z[3][q] - mean) * rs * gg[3 * HDIM + h] + gb[3 * HDIM + h];
        float gv = tanhf(jv);
        float cp = c_state[(size_t)b * HDIM + h];
        float cc = cp * sigm(fv + 1.0f) + sigm(iv) * gv;   // FORGET_BIAS = 1
        cn[q] = cc; ov[q] = oo; csum += cc; csq += cc * cc;
    }
    s1[t] = csum; s2[t] = csq; __syncthreads();
    for (int off = 128; off > 0; off >>= 1) {
        if (t < off) { s1[t] += s1[t + off]; s2[t] += s2[t + off]; }
        __syncthreads();
    }
    float cmean = s1[0] * (1.0f / HDIM);
    float cvar  = s2[0] * (1.0f / HDIM) - cmean * cmean;
    float crs   = rsqrtf(cvar + 1e-5f);

#pragma unroll
    for (int q = 0; q < 4; ++q) {
        int h = t * 4 + q;
        float cl = (cn[q] - cmean) * crs * cg[h] + cb[h];
        float hv = tanhf(cl) * sigm(ov[q]);
        size_t idx = (size_t)b * HDIM + h;
        c_state[idx] = cl;
        h_state[idx] = hv;
        bf16_t hi = (bf16_t)hv;
        h_hi[idx] = hi;
        h_lo[idx] = (bf16_t)(hv - (float)hi);
    }
}

// ---------------------------------------------------------------------------
extern "C" void kernel_launch(void* const* d_in, const int* in_sizes, int n_in,
                              void* d_out, int out_size, void* d_ws, size_t ws_size,
                              hipStream_t stream) {
    const float* x  = (const float*)d_in[0];   // (S,B,I)
    const float* Wx = (const float*)d_in[1];   // (4H,I)
    const float* Wh = (const float*)d_in[2];   // (4H,H)
    const float* bb = (const float*)d_in[3];   // (4H)
    const float* gg = (const float*)d_in[4];   // (4,H)
    const float* gb = (const float*)d_in[5];
    const float* cg = (const float*)d_in[6];   // (H)
    const float* cb = (const float*)d_in[7];

    // Carve workspace (256B aligned).
    char* p = (char*)d_ws;
    auto alloc = [&](size_t bytes) -> char* {
        char* r = p; p += (bytes + 255) & ~(size_t)255; return r;
    };
    const size_t SB = (size_t)S_LEN * BATCH;                 // 32768
    bf16_t* xh  = (bf16_t*)alloc(SB * IDIM * sizeof(bf16_t));
    bf16_t* xl  = (bf16_t*)alloc(SB * IDIM * sizeof(bf16_t));
    bf16_t* Wxh = (bf16_t*)alloc((size_t)G4H * IDIM * sizeof(bf16_t));
    bf16_t* Wxl = (bf16_t*)alloc((size_t)G4H * IDIM * sizeof(bf16_t));
    bf16_t* Whh = (bf16_t*)alloc((size_t)G4H * HDIM * sizeof(bf16_t));
    bf16_t* Whl = (bf16_t*)alloc((size_t)G4H * HDIM * sizeof(bf16_t));
    float*  Xg  = (float*) alloc(SB * G4H * sizeof(float));  // 512 MiB
    float*  Hg  = (float*) alloc((size_t)BATCH * G4H * sizeof(float));
    float*  hs  = (float*) alloc((size_t)BATCH * HDIM * sizeof(float));
    float*  cs  = (float*) alloc((size_t)BATCH * HDIM * sizeof(float));
    bf16_t* hhi = (bf16_t*)alloc((size_t)BATCH * HDIM * sizeof(bf16_t));
    bf16_t* hlo = (bf16_t*)alloc((size_t)BATCH * HDIM * sizeof(bf16_t));

    // 1) bf16 hi/lo splits (one-shot, parallel).
    {
        int n;
        n = (int)(SB * IDIM);
        split_kernel<<<(n + 255) / 256, 256, 0, stream>>>(x, xh, xl, n);
        n = G4H * IDIM;
        split_kernel<<<(n + 255) / 256, 256, 0, stream>>>(Wx, Wxh, Wxl, n);
        n = G4H * HDIM;
        split_kernel<<<(n + 255) / 256, 256, 0, stream>>>(Wh, Whh, Whl, n);
    }

    // 2) Hoisted input projection: Xg = x * Wx^T for all S at once.
    gemm_x_kernel<<<dim3(SB / 16, G4H / 256), 128, 0, stream>>>(xh, xl, Wxh, Wxl, Xg);

    // 3) Zero initial state.
    init_state_kernel<<<(BATCH * HDIM) / 256, 256, 0, stream>>>(hs, cs, hhi, hlo);

    // 4) Sequential recurrence: tiny WMMA GEMM + fused LN cell per step.
    for (int s = 0; s < S_LEN; ++s) {
        gemm_h_kernel<<<128, 128, 0, stream>>>(hhi, hlo, Whh, Whl, Hg);
        cell_kernel<<<BATCH, 256, 0, stream>>>(Xg + (size_t)s * BATCH * G4H, Hg,
                                               bb, gg, gb, cg, cb,
                                               hs, cs, hhi, hlo);
    }

    // 5) Outputs: (h, c) concatenated.
    const size_t bytes = (size_t)BATCH * HDIM * sizeof(float);
    hipMemcpyAsync(d_out, hs, bytes, hipMemcpyDeviceToDevice, stream);
    hipMemcpyAsync((char*)d_out + bytes, cs, bytes, hipMemcpyDeviceToDevice, stream);
}
